// EdgeConvNet_8134668059110
// MI455X (gfx1250) — matrix operand order repaired and checked by reading, not compiled, not run on hardware
//
#include <hip/hip_runtime.h>
#include <hip/hip_bf16.h>
#include <math.h>

// ---------------------------------------------------------------------------
// Types for CDNA5 WMMA (wave32): bf16 A/B fragments (16 elems/lane), f32 acc.
// ---------------------------------------------------------------------------
typedef __bf16 bf16;
typedef __attribute__((ext_vector_type(16))) __bf16 v16bf;
typedef __attribute__((ext_vector_type(8)))  __bf16 bf16x8;
typedef __attribute__((ext_vector_type(8)))  float  v8f;

#define BN_EPS 1e-5f

// ---------------------------------------------------------------------------
// Utility kernels
// ---------------------------------------------------------------------------
__global__ __launch_bounds__(256) void k_zero_f(float* __restrict__ p, unsigned n) {
  unsigned i = blockIdx.x * blockDim.x + threadIdx.x;
  if (i < n) p[i] = 0.f;
}
__global__ __launch_bounds__(256) void k_zero_u(unsigned* __restrict__ p, unsigned n) {
  unsigned i = blockIdx.x * blockDim.x + threadIdx.x;
  if (i < n) p[i] = 0u;
}

// Convert a (K x N) fp32 weight to bf16 **transposed** to [N][Kp] layout
// (zero-padding K up to Kp) so the GEMM stages B tiles with contiguous 16B copies.
__global__ __launch_bounds__(256) void k_w2bfT(const float* __restrict__ W,
                                               bf16* __restrict__ WbT,
                                               int K, int N, int Kp) {
  int i = blockIdx.x * blockDim.x + threadIdx.x;
  if (i >= N * Kp) return;
  int n = i / Kp, k = i % Kp;
  WbT[i] = (k < K) ? (bf16)W[(size_t)k * N + n] : (bf16)0.f;
}

// Conv1 only: materialize per-edge rows [ X[dst], X[src]-X[dst] ] padded to 32.
__global__ __launch_bounds__(256) void k_gather(const float* __restrict__ X,
                                                const int* __restrict__ src,
                                                const int* __restrict__ dst,
                                                bf16* __restrict__ Fe,
                                                unsigned total, int d, int kps) {
  unsigned i = blockIdx.x * blockDim.x + threadIdx.x;
  if (i >= total) return;
  unsigned e = i >> kps;
  int c = (int)(i & ((1u << kps) - 1u));
  float v = 0.f;
  if (c < 2 * d) {
    int sj = src[e], ti = dst[e];
    if (c < d) {
      v = X[(size_t)ti * d + c];
    } else {
      int cc = c - d;
      v = X[(size_t)sj * d + cc] - X[(size_t)ti * d + cc];
    }
  }
  Fe[i] = (bf16)v;
}

// ---------------------------------------------------------------------------
// Shared WMMA tile machinery.
// Block: 256 threads = 8 waves arranged 4(M) x 2(N) -> 64x64 output tile;
// each wave computes a 16x32 slab (2 WMMAs per 32-k chunk). LDS tiles are
// double-buffered: one __syncthreads per chunk.
// Fragment layouts used here (contiguous k, same order for A and B):
//   A (16x32 bf16): lanes 0-15 -> M=lane, K=0..15;
//                   lanes 16-31 -> M=lane-16, K=16..31
//   B (32x16 bf16): lanes 0-15 -> N=lane, K=0..15; lanes 16-31 -> K=16..31
// ---------------------------------------------------------------------------
__device__ __forceinline__ v16bf cat8(bf16x8 lo, bf16x8 hi) {
  return __builtin_shufflevector(lo, hi, 0, 1, 2, 3, 4, 5, 6, 7,
                                 8, 9, 10, 11, 12, 13, 14, 15);
}

// Epilogue: store Y tile (+bias) and optionally reduce per-column sum/sumsq
// (bias included) into global stats via an LDS reduction.
__device__ __forceinline__ void gemm_epilogue(
    v8f acc0, v8f acc1, const float* __restrict__ bias, float* __restrict__ Y,
    float* __restrict__ gsum, float* __restrict__ gssq, int N, int bm, int bn,
    int wm, int wn, int lane, int tid, float* csum, float* cssq) {
  const int n0 = bn + wn * 32 + (lane & 15);
  const int m0 = bm + wm * 16 + (lane >> 4) * 8;
  const float bb0 = bias ? bias[n0] : 0.f;
  const float bb1 = bias ? bias[n0 + 16] : 0.f;
  float s0 = 0.f, q0 = 0.f, s1 = 0.f, q1 = 0.f;
#pragma unroll
  for (int v = 0; v < 8; ++v) {
    float y0 = acc0[v] + bb0;
    float y1 = acc1[v] + bb1;
    Y[(size_t)(m0 + v) * N + n0]      = y0;
    Y[(size_t)(m0 + v) * N + n0 + 16] = y1;
    s0 += y0; q0 += y0 * y0;
    s1 += y1; q1 += y1 * y1;
  }
  if (gsum) {
    if (tid < 64) { csum[tid] = 0.f; cssq[tid] = 0.f; }
    __syncthreads();
    const int c0 = wn * 32 + (lane & 15);
    atomicAdd(&csum[c0], s0);       atomicAdd(&cssq[c0], q0);
    atomicAdd(&csum[c0 + 16], s1);  atomicAdd(&cssq[c0 + 16], q1);
    __syncthreads();
    if (tid < 64) {
      atomicAdd(&gsum[bn + tid], csum[tid]);
      atomicAdd(&gssq[bn + tid], cssq[tid]);
    }
  }
}

// GEMM with pre-materialized bf16 A: Y = A @ B + bias (B given as BT: N x Kp).
__global__ __launch_bounds__(256) void k_wmma_gemm(
    const bf16* __restrict__ A, const bf16* __restrict__ BT,
    const float* __restrict__ bias, float* __restrict__ Y,
    float* __restrict__ gsum, float* __restrict__ gssq, int M, int N, int K) {
  __shared__ bf16 As[2][64][32];
  __shared__ bf16 Bs[2][64][32];
  __shared__ float csum[64], cssq[64];

  const int tid = threadIdx.x, lane = tid & 31, wave = tid >> 5;
  const int wm = wave >> 1, wn = wave & 1;
  const int bm = blockIdx.x * 64, bn = blockIdx.y * 64;

  const int sr = (tid * 8) >> 5;   // 0..63
  const int sc = (tid * 8) & 31;   // {0,8,16,24}
  const bf16* Ag = A  + (size_t)(bm + sr) * K + sc;
  const bf16* Bg = BT + (size_t)(bn + sr) * K + sc;

  v8f acc0 = {0.f, 0.f, 0.f, 0.f, 0.f, 0.f, 0.f, 0.f};
  v8f acc1 = {0.f, 0.f, 0.f, 0.f, 0.f, 0.f, 0.f, 0.f};

  const int arow = wm * 16 + (lane & 15);
  const int ka   = (lane >> 4) * 16;
  const int bn0  = wn * 32 + (lane & 15);
  const int bn1  = bn0 + 16;
  const int kb   = (lane >> 4) * 16;

  *(bf16x8*)&As[0][sr][sc] = *(const bf16x8*)(Ag);
  *(bf16x8*)&Bs[0][sr][sc] = *(const bf16x8*)(Bg);
  __syncthreads();

  const int nk = K >> 5;
  for (int j = 0; j < nk; ++j) {
    const int buf = j & 1;
    if (j + 1 < nk) {
      *(bf16x8*)&As[buf ^ 1][sr][sc] = *(const bf16x8*)(Ag + ((j + 1) << 5));
      *(bf16x8*)&Bs[buf ^ 1][sr][sc] = *(const bf16x8*)(Bg + ((j + 1) << 5));
    }
    v16bf av  = cat8(*(const bf16x8*)&As[buf][arow][ka],
                     *(const bf16x8*)&As[buf][arow][ka + 8]);
    v16bf bv0 = cat8(*(const bf16x8*)&Bs[buf][bn0][kb],
                     *(const bf16x8*)&Bs[buf][bn0][kb + 8]);
    v16bf bv1 = cat8(*(const bf16x8*)&Bs[buf][bn1][kb],
                     *(const bf16x8*)&Bs[buf][bn1][kb + 8]);
    acc0 = __builtin_amdgcn_wmma_f32_16x16x32_bf16(false, av, false, bv0,
                                                   (short)0, acc0, false, false);
    acc1 = __builtin_amdgcn_wmma_f32_16x16x32_bf16(false, av, false, bv1,
                                                   (short)0, acc1, false, false);
    __syncthreads();
  }
  gemm_epilogue(acc0, acc1, bias, Y, gsum, gssq, N, bm, bn, wm, wn, lane, tid,
                csum, cssq);
}

// Gather-fused GEMM for EdgeConv: A row e = [ H[dst[e]], H[src[e]]-H[dst[e]] ]
// built on the fly from the (L2-resident) node-feature table H (Nn x d, f32).
// K = 2d, d multiple of 32 -> no chunk straddles the concat boundary.
__global__ __launch_bounds__(256) void k_wmma_gemm_edge(
    const float* __restrict__ H, const int* __restrict__ src,
    const int* __restrict__ dst, const bf16* __restrict__ BT,
    const float* __restrict__ bias, float* __restrict__ Y,
    float* __restrict__ gsum, float* __restrict__ gssq, int M, int N, int K,
    int d) {
  __shared__ bf16 As[2][64][32];
  __shared__ bf16 Bs[2][64][32];
  __shared__ float csum[64], cssq[64];

  const int tid = threadIdx.x, lane = tid & 31, wave = tid >> 5;
  const int wm = wave >> 1, wn = wave & 1;
  const int bm = blockIdx.x * 64, bn = blockIdx.y * 64;

  const int sr = (tid * 8) >> 5;
  const int sc = (tid * 8) & 31;
  const int e  = bm + sr;              // edge id staged by this thread
  const int sj = src[e], ti = dst[e];  // k-invariant: load once
  const float* __restrict__ Hs = H + (size_t)sj * d;
  const float* __restrict__ Ht = H + (size_t)ti * d;
  const bf16* Bg = BT + (size_t)(bn + sr) * K + sc;

  v8f acc0 = {0.f, 0.f, 0.f, 0.f, 0.f, 0.f, 0.f, 0.f};
  v8f acc1 = {0.f, 0.f, 0.f, 0.f, 0.f, 0.f, 0.f, 0.f};

  const int arow = wm * 16 + (lane & 15);
  const int ka   = (lane >> 4) * 16;
  const int bn0  = wn * 32 + (lane & 15);
  const int bn1  = bn0 + 16;
  const int kb   = (lane >> 4) * 16;

  auto stageA = [&](int buf, int c0) {
    float v[8];
    if (c0 < d) {
      const float4 f0 = *(const float4*)(Ht + c0);
      const float4 f1 = *(const float4*)(Ht + c0 + 4);
      v[0] = f0.x; v[1] = f0.y; v[2] = f0.z; v[3] = f0.w;
      v[4] = f1.x; v[5] = f1.y; v[6] = f1.z; v[7] = f1.w;
    } else {
      const int cc = c0 - d;
      const float4 s0 = *(const float4*)(Hs + cc);
      const float4 s1 = *(const float4*)(Hs + cc + 4);
      const float4 t0 = *(const float4*)(Ht + cc);
      const float4 t1 = *(const float4*)(Ht + cc + 4);
      v[0] = s0.x - t0.x; v[1] = s0.y - t0.y; v[2] = s0.z - t0.z;
      v[3] = s0.w - t0.w; v[4] = s1.x - t1.x; v[5] = s1.y - t1.y;
      v[6] = s1.z - t1.z; v[7] = s1.w - t1.w;
    }
    bf16x8 o;
#pragma unroll
    for (int i = 0; i < 8; ++i) o[i] = (bf16)v[i];
    *(bf16x8*)&As[buf][sr][sc] = o;
  };

  stageA(0, sc);
  *(bf16x8*)&Bs[0][sr][sc] = *(const bf16x8*)(Bg);
  __syncthreads();

  const int nk = K >> 5;
  for (int j = 0; j < nk; ++j) {
    const int buf = j & 1;
    if (j + 1 < nk) {
      stageA(buf ^ 1, ((j + 1) << 5) + sc);
      *(bf16x8*)&Bs[buf ^ 1][sr][sc] = *(const bf16x8*)(Bg + ((j + 1) << 5));
    }
    v16bf av  = cat8(*(const bf16x8*)&As[buf][arow][ka],
                     *(const bf16x8*)&As[buf][arow][ka + 8]);
    v16bf bv0 = cat8(*(const bf16x8*)&Bs[buf][bn0][kb],
                     *(const bf16x8*)&Bs[buf][bn0][kb + 8]);
    v16bf bv1 = cat8(*(const bf16x8*)&Bs[buf][bn1][kb],
                     *(const bf16x8*)&Bs[buf][bn1][kb + 8]);
    acc0 = __builtin_amdgcn_wmma_f32_16x16x32_bf16(false, av, false, bv0,
                                                   (short)0, acc0, false, false);
    acc1 = __builtin_amdgcn_wmma_f32_16x16x32_bf16(false, av, false, bv1,
                                                   (short)0, acc1, false, false);
    __syncthreads();
  }
  gemm_epilogue(acc0, acc1, bias, Y, gsum, gssq, N, bm, bn, wm, wn, lane, tid,
                csum, cssq);
}

// s = g * rsqrt(var + eps);  t = beta - mu * s   (biased batch variance)
__global__ void k_bnfin(const float* __restrict__ sum,
                        const float* __restrict__ sumsq,
                        const float* __restrict__ g,
                        const float* __restrict__ be,
                        float* __restrict__ s, float* __restrict__ t, int M) {
  int c = threadIdx.x;
  float mu  = sum[c] / (float)M;
  float var = sumsq[c] / (float)M - mu * mu;
  float sc  = g[c] * rsqrtf(var + BN_EPS);
  s[c] = sc;
  t[c] = be[c] - mu * sc;
}

// out = relu(Y * s[col] + t[col]); writes f32 and/or bf16. N = 1<<ns.
__global__ __launch_bounds__(256) void k_scale_bias_relu(
    const float* __restrict__ Y, const float* __restrict__ s,
    const float* __restrict__ t, float* __restrict__ outF,
    bf16* __restrict__ outB, unsigned total, int ns) {
  unsigned i = blockIdx.x * blockDim.x + threadIdx.x;
  if (i >= total) return;
  int c = (int)(i & ((1u << ns) - 1u));
  float ss = s ? s[c] : 1.f;
  float tt = t ? t[c] : 0.f;
  float v = fmaxf(Y[i] * ss + tt, 0.f);
  if (outF) outF[i] = v;
  if (outB) outB[i] = (bf16)v;
}

// Fused BN-ReLU + segment_max: v = relu(Y*s + t) folded straight into the
// per-node uint-bit atomic max (values >= 0 post-ReLU; init-0 reproduces the
// reference "empty node -> 0"). Eliminates the fp32 relu write-back and the
// separate aggregation read pass. d = 1<<ds_.
__global__ __launch_bounds__(256) void k_sbrelu_max(
    const float* __restrict__ Y, const float* __restrict__ s,
    const float* __restrict__ t, const int* __restrict__ dst,
    unsigned* __restrict__ H, unsigned total, int ds_) {
  unsigned i = blockIdx.x * blockDim.x + threadIdx.x;
  if (i >= total) return;
  unsigned e = i >> ds_;
  unsigned c = i & ((1u << ds_) - 1u);
  float v = fmaxf(Y[i] * s[c] + t[c], 0.f);
  atomicMax(H + (((size_t)dst[e]) << ds_) + c, __float_as_uint(v));
}

// Per-graph sum / max / count accumulation over nodes. d = 1<<ds_.
__global__ __launch_bounds__(256) void k_pool_acc(const float* __restrict__ H,
                                                  const int* __restrict__ batch,
                                                  float* __restrict__ psum,
                                                  unsigned* __restrict__ pmax,
                                                  float* __restrict__ cnt,
                                                  unsigned total, int ds_) {
  unsigned i = blockIdx.x * blockDim.x + threadIdx.x;
  if (i >= total) return;
  unsigned n = i >> ds_;
  unsigned c = i & ((1u << ds_) - 1u);
  int b = batch[n];
  float v = H[i];
  atomicAdd(&psum[(((size_t)b) << ds_) + c], v);
  atomicMax(&pmax[(((size_t)b) << ds_) + c], __float_as_uint(v));
  if (c == 0) atomicAdd(&cnt[b], 1.f);
}

// p[b] = [ psum/max(cnt,1) | max ] -> (64 x 512), f32 + bf16 copies.
__global__ __launch_bounds__(256) void k_pool_finalize(
    const float* __restrict__ psum, const unsigned* __restrict__ pmax,
    const float* __restrict__ cnt, float* __restrict__ p,
    bf16* __restrict__ pb) {
  int i = blockIdx.x * blockDim.x + threadIdx.x;
  if (i >= 64 * 512) return;
  int b = i >> 9, c = i & 511;
  float v = (c < 256) ? psum[b * 256 + c] / fmaxf(cnt[b], 1.f)
                      : __uint_as_float(pmax[b * 256 + (c - 256)]);
  p[i] = v;
  pb[i] = (bf16)v;
}

// Final 64 -> 2 linear + log_softmax, one thread per graph.
__global__ void k_final(const float* __restrict__ act,
                        const float* __restrict__ W,
                        const float* __restrict__ b, float* __restrict__ out) {
  int r = threadIdx.x;  // 0..63
  float z0 = b[0], z1 = b[1];
  for (int k = 0; k < 64; ++k) {
    float a = act[r * 64 + k];
    z0 += a * W[k * 2 + 0];
    z1 += a * W[k * 2 + 1];
  }
  float m = fmaxf(z0, z1);
  float l = m + logf(expf(z0 - m) + expf(z1 - m));
  out[r * 2 + 0] = z0 - l;
  out[r * 2 + 1] = z1 - l;
}

// ---------------------------------------------------------------------------
// Host orchestration
// ---------------------------------------------------------------------------
extern "C" void kernel_launch(void* const* d_in, const int* in_sizes, int n_in,
                              void* d_out, int out_size, void* d_ws,
                              size_t ws_size, hipStream_t stream) {
  (void)n_in; (void)out_size; (void)ws_size;
  const int Nn = in_sizes ? in_sizes[2] : 10000;       // nodes
  const int E  = in_sizes ? in_sizes[1] / 2 : 320000;  // edges

  const float* x    = (const float*)d_in[0];
  const int*  ei    = (const int*)d_in[1];
  const int*  src   = ei;       // edge_index[0] = j
  const int*  dst   = ei + E;   // edge_index[1] = i
  const int*  batch = (const int*)d_in[2];

  const float *W11 = (const float*)d_in[3],  *b11 = (const float*)d_in[4];
  const float *g11 = (const float*)d_in[5],  *be11 = (const float*)d_in[6];
  const float *W12 = (const float*)d_in[7],  *b12 = (const float*)d_in[8];
  const float *g12 = (const float*)d_in[9],  *be12 = (const float*)d_in[10];
  const float *W21 = (const float*)d_in[11], *b21 = (const float*)d_in[12];
  const float *g21 = (const float*)d_in[13], *be21 = (const float*)d_in[14];
  const float *W22 = (const float*)d_in[15], *b22 = (const float*)d_in[16];
  const float *g22 = (const float*)d_in[17], *be22 = (const float*)d_in[18];
  const float *W31 = (const float*)d_in[19], *b31 = (const float*)d_in[20];
  const float *g31 = (const float*)d_in[21], *be31 = (const float*)d_in[22];
  const float *Wc1 = (const float*)d_in[23], *bc1 = (const float*)d_in[24];
  const float *gc1 = (const float*)d_in[25], *bec1 = (const float*)d_in[26];
  const float *Wc2 = (const float*)d_in[27], *bc2 = (const float*)d_in[28];
  const float *Wc3 = (const float*)d_in[29], *bc3 = (const float*)d_in[30];
  float* out = (float*)d_out;

  // Workspace carve-out
  char* ws = (char*)d_ws;
  size_t off = 0;
  auto carve = [&](size_t bytes) {
    size_t o = off;
    off += (bytes + 255) & ~(size_t)255;
    return o;
  };
  bf16*  Abuf = (bf16*)(ws + carve((size_t)E * 32 * sizeof(bf16)));    // conv1 A
  float* Ybuf = (float*)(ws + carve((size_t)E * 256 * sizeof(float))); // GEMM out
  bf16*  Actb = (bf16*)(ws + carve((size_t)E * 128 * sizeof(bf16)));   // bf16 act
  float* H1   = (float*)(ws + carve((size_t)Nn * 64 * sizeof(float)));
  float* H2   = (float*)(ws + carve((size_t)Nn * 128 * sizeof(float)));
  float* H3   = (float*)(ws + carve((size_t)Nn * 256 * sizeof(float)));
  bf16*  WbT  = (bf16*)(ws + carve((size_t)512 * 256 * sizeof(bf16)));
  float* stat = (float*)(ws + carve(1024 * sizeof(float)));  // sum|sq|s|t
  float* psum = (float*)(ws + carve(64 * 256 * sizeof(float)));
  unsigned* pmax = (unsigned*)(ws + carve(64 * 256 * sizeof(unsigned)));
  float* cnt  = (float*)(ws + carve(64 * sizeof(float)));
  float* pbuf = (float*)(ws + carve(64 * 512 * sizeof(float)));
  bf16*  pb   = (bf16*)(ws + carve(64 * 512 * sizeof(bf16)));

  float* gsum = stat;        // [0,256)
  float* gssq = stat + 256;  // [256,512)
  float* sb   = stat + 512;
  float* tb   = stat + 768;

  auto cdiv = [](size_t a, size_t b) { return (unsigned)((a + b - 1) / b); };
  auto zstats = [&]() { k_zero_f<<<2, 256, 0, stream>>>(gsum, 512); };

  // GEMM on pre-materialized bf16 A (with optional fused BN stats).
  auto gemm = [&](const bf16* A, const float* W, const float* bias, float* Y,
                  int M, int N, int K, int Kp, bool stats) {
    k_w2bfT<<<cdiv((size_t)N * Kp, 256), 256, 0, stream>>>(W, WbT, K, N, Kp);
    dim3 grid((unsigned)(M / 64), (unsigned)(N / 64));
    k_wmma_gemm<<<grid, 256, 0, stream>>>(A, WbT, bias, Y,
                                          stats ? gsum : nullptr,
                                          stats ? gssq : nullptr, M, N, Kp);
  };
  // Gather-fused GEMM over edges (A built from node features H, K = 2d).
  auto gemm_edge = [&](const float* H, const float* W, const float* bias,
                       float* Y, int N, int d) {
    const int K = 2 * d;
    k_w2bfT<<<cdiv((size_t)N * K, 256), 256, 0, stream>>>(W, WbT, K, N, K);
    dim3 grid((unsigned)(E / 64), (unsigned)(N / 64));
    k_wmma_gemm_edge<<<grid, 256, 0, stream>>>(H, src, dst, WbT, bias, Y,
                                               gsum, gssq, E, N, K, d);
  };
  // BN finalize + ReLU feeding the next GEMM (bf16 activation output).
  auto bnfin_relu = [&](float* Y, const float* g_, const float* be_, int M,
                        int N, int nshift, bf16* outB) {
    k_bnfin<<<1, N, 0, stream>>>(gsum, gssq, g_, be_, sb, tb, M);
    unsigned tot = (unsigned)M << nshift;
    k_scale_bias_relu<<<cdiv(tot, 256), 256, 0, stream>>>(Y, sb, tb, nullptr,
                                                          outB, tot, nshift);
  };
  // BN finalize + ReLU fused into segment-max aggregation (conv tail).
  auto bnfin_relu_max = [&](float* Y, const float* g_, const float* be_,
                            int N, int nshift, float* Hout) {
    k_bnfin<<<1, N, 0, stream>>>(gsum, gssq, g_, be_, sb, tb, E);
    k_zero_u<<<cdiv(((size_t)Nn) << nshift, 256), 256, 0, stream>>>(
        (unsigned*)Hout, (unsigned)Nn << nshift);
    unsigned tot = (unsigned)E << nshift;
    k_sbrelu_max<<<cdiv(tot, 256), 256, 0, stream>>>(Y, sb, tb, dst,
                                                     (unsigned*)Hout, tot,
                                                     nshift);
  };

  // ---------------- EdgeConv 1: 5 -> 64 -> 64 ----------------
  k_gather<<<cdiv((size_t)E * 32, 256), 256, 0, stream>>>(
      x, src, dst, Abuf, (unsigned)E * 32u, 5, 5);
  zstats();
  gemm(Abuf, W11, b11, Ybuf, E, 64, 10, 32, true);
  bnfin_relu(Ybuf, g11, be11, E, 64, 6, Actb);
  zstats();
  gemm(Actb, W12, b12, Ybuf, E, 64, 64, 64, true);
  bnfin_relu_max(Ybuf, g12, be12, 64, 6, H1);

  // ---------------- EdgeConv 2: 64 -> 128 -> 128 ----------------
  zstats();
  gemm_edge(H1, W21, b21, Ybuf, 128, 64);
  bnfin_relu(Ybuf, g21, be21, E, 128, 7, Actb);
  zstats();
  gemm(Actb, W22, b22, Ybuf, E, 128, 128, 128, true);
  bnfin_relu_max(Ybuf, g22, be22, 128, 7, H2);

  // ---------------- EdgeConv 3: 128 -> 256 ----------------
  zstats();
  gemm_edge(H2, W31, b31, Ybuf, 256, 128);
  bnfin_relu_max(Ybuf, g31, be31, 256, 8, H3);

  // ---------------- Global mean+max pool ----------------
  k_zero_f<<<cdiv(64 * 256, 256), 256, 0, stream>>>(psum, 64u * 256u);
  k_zero_u<<<cdiv(64 * 256, 256), 256, 0, stream>>>(pmax, 64u * 256u);
  k_zero_f<<<1, 64, 0, stream>>>(cnt, 64u);
  k_pool_acc<<<cdiv((size_t)Nn * 256, 256), 256, 0, stream>>>(
      H3, batch, psum, pmax, cnt, (unsigned)Nn * 256u, 8);
  k_pool_finalize<<<cdiv(64 * 512, 256), 256, 0, stream>>>(psum, pmax, cnt,
                                                           pbuf, pb);

  // ---------------- Classifier head ----------------
  zstats();
  gemm(pb, Wc1, bc1, Ybuf, 64, 256, 512, 512, true);
  bnfin_relu(Ybuf, gc1, bec1, 64, 256, 8, Actb);
  gemm(Actb, Wc2, bc2, Ybuf, 64, 64, 256, 256, false);
  k_scale_bias_relu<<<cdiv(64 * 64, 256), 256, 0, stream>>>(
      Ybuf, nullptr, nullptr, Ybuf, nullptr, 64u * 64u, 6);
  k_final<<<1, 64, 0, stream>>>(Ybuf, Wc3, bc3, out);
}
